// GL_GPT2Layer_56006373540309
// MI455X (gfx1250) — matrix operand order repaired and checked by reading, not compiled, not run on hardware
//
#include <hip/hip_runtime.h>
#include <hip/hip_bf16.h>
#include <math.h>
#include <stdint.h>

// GPT-2 layer dims
#define BB 2
#define SS 2048
#define EE 1024
#define HH 16
#define DDIM 64
#define MROWS (BB*SS)   // 4096

typedef __attribute__((ext_vector_type(4)))  __bf16 v4bf;
typedef __attribute__((ext_vector_type(8)))  __bf16 v8bf;
typedef __attribute__((ext_vector_type(16))) __bf16 v16bf;
typedef __attribute__((ext_vector_type(8)))  float  v8f;

// ---- CDNA5 async / TDM feature detection (compile-safe fallbacks) ----------
#ifdef __has_builtin
#if __has_builtin(__builtin_amdgcn_global_load_async_to_lds_b128)
#define HAVE_ASYNC_LDS 1
#endif
#if __has_builtin(__builtin_amdgcn_tensor_load_to_lds)
#define HAVE_TDM 1
#endif
#endif

// Builtin expects pointers to GCC-style vector_size(16) int, AS1 / AS3.
typedef int vsi4 __attribute__((vector_size(16)));
typedef __attribute__((address_space(1))) vsi4 gm_i4;
typedef __attribute__((address_space(3))) vsi4 lds_i4;

static __device__ __forceinline__ gm_i4* to_gm4(const void* p) {
  return (gm_i4*)(unsigned long long)p;
}
static __device__ __forceinline__ lds_i4* to_lds4(const void* p) {
  // generic LDS address truncates to LDS offset (ISA aperture rule)
  return (lds_i4*)(unsigned)(unsigned long long)p;
}
static __device__ __forceinline__ unsigned lds_off(const void* p) {
  return (unsigned)(unsigned long long)p;
}

static __device__ __forceinline__ void wait_async0() {
#ifdef __has_builtin
#if __has_builtin(__builtin_amdgcn_s_wait_asynccnt)
  __builtin_amdgcn_s_wait_asynccnt(0);
#else
  asm volatile("s_wait_asynccnt 0x0" ::: "memory");
#endif
#else
  asm volatile("s_wait_asynccnt 0x0" ::: "memory");
#endif
}

#if defined(HAVE_TDM)
// TDM 2D bf16 tile load: tile_rows x tile_d elems, row stride in elems,
// optional LDS row padding so the DMA lands in a padded LDS array.
static __device__ __forceinline__ void tdm_load_2d_bf16(
    const __bf16* gsrc, unsigned lds_byte_off,
    int tile_d, int tile_rows, int rows_avail, int row_stride_elems,
    int pad_interval_code, int pad_amount_code, int pad_enable) {
  typedef __attribute__((ext_vector_type(4))) unsigned int v4u_;
  typedef __attribute__((ext_vector_type(4))) int v4i_;
  typedef __attribute__((ext_vector_type(8))) int v8i_;
  const unsigned long long ga = (unsigned long long)gsrc;
  v4u_ g0;
  g0[0] = 1u;                                               // count=1 (valid user D#)
  g0[1] = lds_byte_off;                                     // lds_addr
  g0[2] = (unsigned)(ga & 0xffffffffu);                     // global_addr[31:0]
  g0[3] = (unsigned)((ga >> 32) & 0x01ffffffu) | (2u << 30);// global_addr[56:32]|type=2
  v8i_ g1;
  g1[0] = (1 << 16) | (pad_enable << 20) |                  // data_size=2B, pad ctrl
          (pad_interval_code << 22) | (pad_amount_code << 25);
  g1[1] = (tile_d & 0xffff) << 16;                          // tensor_dim0 lo16
  g1[2] = ((unsigned)rows_avail & 0xffffu) << 16;           // dim0 hi16=0 | tensor_dim1 lo16
  g1[3] = (int)((((unsigned)rows_avail >> 16) & 0xffffu) |  // tensor_dim1 hi16
                ((unsigned)(tile_d & 0xffff) << 16));       // tile_dim0
  g1[4] = tile_rows & 0xffff;                               // tile_dim1 | tile_dim2=0
  g1[5] = row_stride_elems;                                 // tensor_dim0_stride lo32
  g1[6] = 0;
  g1[7] = 0;
  v4i_ gz = {0, 0, 0, 0};
#if __clang_major__ >= 23
  v8i_ gz8 = {0, 0, 0, 0, 0, 0, 0, 0};
  __builtin_amdgcn_tensor_load_to_lds(g0, g1, gz, gz, gz8, 0);
#else
  __builtin_amdgcn_tensor_load_to_lds(g0, g1, gz, gz, 0);
#endif
}
#endif

static __device__ __forceinline__ v16bf cat16(v8bf a, v8bf b) {
  return __builtin_shufflevector(a, b, 0,1,2,3,4,5,6,7,8,9,10,11,12,13,14,15);
}

static __device__ __forceinline__ v8f wmma_bf16(v16bf a, v16bf b, v8f c) {
  // D = A(16x32 bf16) * B(32x16 bf16) + C(16x16 f32)
  return __builtin_amdgcn_wmma_f32_16x16x32_bf16(false, a, false, b, (short)0, c, false, false);
}

static __device__ __forceinline__ float gelu_new_f(float x) {
  const float c = 0.7978845608028654f; // sqrt(2/pi)
  return 0.5f * x * (1.0f + tanhf(c * (x + 0.044715f * x * x * x)));
}

// ---------------------------------------------------------------------------
// fp32 -> bf16 conversion (weights), vectorized x4
// ---------------------------------------------------------------------------
__global__ void cvt_f32_bf16_kernel(const float* __restrict__ in,
                                    __bf16* __restrict__ out, int n4) {
  int i = blockIdx.x * blockDim.x + threadIdx.x;
  if (i < n4) {
    float4 v = ((const float4*)in)[i];
    v4bf o;
    o[0] = (__bf16)v.x; o[1] = (__bf16)v.y; o[2] = (__bf16)v.z; o[3] = (__bf16)v.w;
    ((v4bf*)out)[i] = o;
  }
}

// ---------------------------------------------------------------------------
// LayerNorm (one row of E=1024 per block of 256 threads) -> bf16
// ---------------------------------------------------------------------------
__global__ void ln_kernel(const float* __restrict__ x, const float* __restrict__ g,
                          const float* __restrict__ bta, __bf16* __restrict__ out) {
  const int row = blockIdx.x;
  const int t = threadIdx.x;           // 0..255, 4 elems each
  const float4 xv = ((const float4*)(x + (size_t)row * EE))[t];

  __shared__ float red[8];
  float s = xv.x + xv.y + xv.z + xv.w;
  #pragma unroll
  for (int m = 16; m >= 1; m >>= 1) s += __shfl_xor(s, m, 32);
  if ((t & 31) == 0) red[t >> 5] = s;
  __syncthreads();
  float mu = 0.f;
  #pragma unroll
  for (int i = 0; i < 8; i++) mu += red[i];
  mu *= (1.0f / (float)EE);
  __syncthreads();

  const float d0 = xv.x - mu, d1 = xv.y - mu, d2 = xv.z - mu, d3 = xv.w - mu;
  float q = d0*d0 + d1*d1 + d2*d2 + d3*d3;
  #pragma unroll
  for (int m = 16; m >= 1; m >>= 1) q += __shfl_xor(q, m, 32);
  if ((t & 31) == 0) red[t >> 5] = q;
  __syncthreads();
  float var = 0.f;
  #pragma unroll
  for (int i = 0; i < 8; i++) var += red[i];
  var *= (1.0f / (float)EE);
  const float rstd = rsqrtf(var + 1e-5f);

  const float4 gv = ((const float4*)g)[t];
  const float4 bv = ((const float4*)bta)[t];
  v4bf o;
  o[0] = (__bf16)(d0 * rstd * gv.x + bv.x);
  o[1] = (__bf16)(d1 * rstd * gv.y + bv.y);
  o[2] = (__bf16)(d2 * rstd * gv.z + bv.z);
  o[3] = (__bf16)(d3 * rstd * gv.w + bv.w);
  ((v4bf*)(out + (size_t)row * EE))[t] = o;
}

// ---------------------------------------------------------------------------
// Tiled WMMA GEMM: C[M,N] = act(A_bf16[M,K] @ W_bf16[K,N] + bias) (+resid)
// Block: 256 threads (8 waves), tile 128x128, BK=32.
// Wave grid 2(M) x 4(N): each wave owns 64 rows x 32 cols = 4x2 wmma tiles.
// A tile staged via GLOBAL_LOAD_ASYNC_TO_LDS_B128 when available.
// ---------------------------------------------------------------------------
template<int ACT, bool RESID, bool OUTBF>
__global__ __launch_bounds__(256)
void gemm_kernel(const __bf16* __restrict__ A, const __bf16* __restrict__ W,
                 const float* __restrict__ bias, const float* __restrict__ resid,
                 float* __restrict__ outF, __bf16* __restrict__ outB,
                 int Mdim, int Ndim, int Kdim) {
  __shared__ __bf16 As[128][40];   // [m][k], +8 pad
  __shared__ __bf16 Bs[128][40];   // transposed: [n][k], +8 pad

  const int tid  = threadIdx.x;
  const int wave = tid >> 5;
  const int lane = tid & 31;
  const int kh   = lane >> 4;      // 0/1 : lane half
  const int lm   = lane & 15;

  const int waveM = wave >> 2;     // 0..1
  const int waveN = wave & 3;      // 0..3
  const int mBase = blockIdx.y * 128;
  const int nBase = blockIdx.x * 128;

  v8f acc[4][2] = {};

  // staging indices (constant across K loop)
  const int arow = tid >> 1;             // 0..127
  const int acol = (tid & 1) * 16;       // 0/16
  const int brow = tid >> 3;             // 0..31 (k within block)
  const int bcol = (tid & 7) * 16;       // 0..112 (n within block)

  for (int kb = 0; kb < Kdim; kb += 32) {
    // ---- stage A tile (128x32) ----
    {
      const __bf16* src = A + (size_t)(mBase + arow) * Kdim + kb + acol;
#if defined(HAVE_ASYNC_LDS)
      __builtin_amdgcn_global_load_async_to_lds_b128(
          to_gm4(src), to_lds4(&As[arow][acol]), 0, 0);
      __builtin_amdgcn_global_load_async_to_lds_b128(
          to_gm4(src + 8), to_lds4(&As[arow][acol + 8]), 0, 0);
#else
      v8bf lo = *(const v8bf*)(src);
      v8bf hi = *(const v8bf*)(src + 8);
      *(v8bf*)(&As[arow][acol])     = lo;
      *(v8bf*)(&As[arow][acol + 8]) = hi;
#endif
      if (kb + 32 < Kdim) __builtin_prefetch(src + 32, 0, 1);
    }
    // ---- stage B tile transposed (W is [K,N] row-major) ----
    {
      const __bf16* src = W + (size_t)(kb + brow) * Ndim + nBase + bcol;
      v8bf w0 = *(const v8bf*)src;
      v8bf w1 = *(const v8bf*)(src + 8);
      #pragma unroll
      for (int i = 0; i < 8; i++) Bs[bcol + i][brow] = w0[i];
      #pragma unroll
      for (int i = 0; i < 8; i++) Bs[bcol + 8 + i][brow] = w1[i];
      if (kb + 32 < Kdim) __builtin_prefetch(src + (size_t)32 * Ndim, 0, 1);
    }
#if defined(HAVE_ASYNC_LDS)
    wait_async0();
#endif
    __syncthreads();

    // ---- B fragments (per wave, 2 n-tiles) ----
    v16bf bfrag[2];
    #pragma unroll
    for (int nt = 0; nt < 2; nt++) {
      const __bf16* p = &Bs[waveN * 32 + nt * 16 + lm][kh * 8];
      bfrag[nt] = cat16(*(const v8bf*)p, *(const v8bf*)(p + 16));
    }
    // ---- A fragments + WMMA ----
    #pragma unroll
    for (int mt = 0; mt < 4; mt++) {
      const int r = waveM * 64 + mt * 16 + lm;
      v8bf a0 = *(const v8bf*)(&As[r][kh * 8]);
      v8bf a1 = *(const v8bf*)(&As[r][16 + kh * 8]);
      v16bf af = cat16(a0, a1);
      #pragma unroll
      for (int nt = 0; nt < 2; nt++)
        acc[mt][nt] = wmma_bf16(af, bfrag[nt], acc[mt][nt]);
    }
    __syncthreads();
  }

  // ---- epilogue: C layout: elem r -> row r+8*kh, col lm ----
  #pragma unroll
  for (int mt = 0; mt < 4; mt++) {
    #pragma unroll
    for (int nt = 0; nt < 2; nt++) {
      const int gcol = nBase + waveN * 32 + nt * 16 + lm;
      const float bv = bias[gcol];
      #pragma unroll
      for (int r = 0; r < 8; r++) {
        const int grow = mBase + waveM * 64 + mt * 16 + r + 8 * kh;
        float v = acc[mt][nt][r] + bv;
        if (ACT == 1) v = gelu_new_f(v);
        if (RESID) v += resid[(size_t)grow * Ndim + gcol];
        if (OUTBF) outB[(size_t)grow * Ndim + gcol] = (__bf16)v;
        else       outF[(size_t)grow * Ndim + gcol] = v;
      }
    }
  }
}

// ---------------------------------------------------------------------------
// Flash attention (causal). qkv: bf16 [B*S, 3E]; Q at +0, K at +E, V at +2E.
// Grid: (S/128, B*H). Block: 256 (8 waves). Wave w handles 16 q-rows.
// K tile staged by the Tensor Data Mover (one descriptor per block-iter)
// when available; V transposed manually; P round-trips via LDS.
// ---------------------------------------------------------------------------
__global__ __launch_bounds__(256)
void attn_kernel(const __bf16* __restrict__ qkv, __bf16* __restrict__ attn_out) {
  __shared__ __bf16 Kt[32][72];        // [kv][d], row stride 144B (TDM padded)
  __shared__ __bf16 Vt[64][40];        // transposed [d][kv]
  __shared__ __bf16 Pb[8][16][40];     // per-wave P (16x32)

  const int tid  = threadIdx.x;
  const int wave = tid >> 5;
  const int lane = tid & 31;
  const int kh   = lane >> 4;
  const int lm   = lane & 15;

  const int bh = blockIdx.y;
  const int bb = bh / HH;
  const int hh = bh % HH;
  const int q0 = blockIdx.x * 128;
  const int qw = q0 + wave * 16;       // wave's first q row

  // Q fragments (held in registers for whole kernel), rows qw+lm, D=64 -> 2 frags
  v16bf qfrag[2];
  {
    const __bf16* qp = qkv + (size_t)(bb * SS + qw + lm) * (3 * EE) + hh * DDIM;
    #pragma unroll
    for (int kk = 0; kk < 2; kk++) {
      v8bf a0 = *(const v8bf*)(qp + kk * 32 + kh * 8);
      v8bf a1 = *(const v8bf*)(qp + kk * 32 + 16 + kh * 8);
      qfrag[kk] = cat16(a0, a1);
    }
  }

  float m_row[8], l_row[8];
  #pragma unroll
  for (int r = 0; r < 8; r++) { m_row[r] = -1e30f; l_row[r] = 0.f; }
  v8f o[4] = {};

  const int jmax = blockIdx.x * 4 + 4;   // kv blocks of 32 covering q0+127
  const int kvr = tid >> 3;              // staging: 0..31
  const int dd0 = (tid & 7) * 8;         // staging: 0..56

  for (int j = 0; j < jmax; j++) {
    const int kv0 = j * 32;
    // ---- stage K ----
#if defined(HAVE_TDM)
    if (wave == 0) {
      const __bf16* ksrc = qkv + (size_t)(bb * SS + kv0) * (3 * EE) + EE + hh * DDIM;
      // 32x64 bf16 tile, row stride 3072 elems; LDS pad 4 DWORDs every 32 DWORDs
      tdm_load_2d_bf16(ksrc, lds_off(&Kt[0][0]), DDIM, 32, SS - kv0, 3 * EE,
                       /*interval 32 dw*/ 4, /*amount 4 dw*/ 3, /*pad*/ 1);
      __builtin_amdgcn_s_wait_tensorcnt(0);
    }
#elif defined(HAVE_ASYNC_LDS)
    {
      const __bf16* ksrc =
          qkv + (size_t)(bb * SS + kv0 + kvr) * (3 * EE) + EE + hh * DDIM + dd0;
      __builtin_amdgcn_global_load_async_to_lds_b128(
          to_gm4(ksrc), to_lds4(&Kt[kvr][dd0]), 0, 0);
    }
#else
    {
      const __bf16* ksrc =
          qkv + (size_t)(bb * SS + kv0 + kvr) * (3 * EE) + EE + hh * DDIM + dd0;
      *(v8bf*)(&Kt[kvr][dd0]) = *(const v8bf*)ksrc;
    }
#endif
    // ---- stage V transposed ----
    {
      const __bf16* vsrc =
          qkv + (size_t)(bb * SS + kv0 + kvr) * (3 * EE) + 2 * EE + hh * DDIM + dd0;
      v8bf vv = *(const v8bf*)vsrc;
      #pragma unroll
      for (int i = 0; i < 8; i++) Vt[dd0 + i][kvr] = vv[i];
    }
#if !defined(HAVE_TDM) && defined(HAVE_ASYNC_LDS)
    wait_async0();
#endif
    __syncthreads();

    // ---- S = (Q K^T) * scale : 2 tiles of 16 kv cols, K-dim = D = 64 ----
    v8f st[2] = {};
    #pragma unroll
    for (int nt = 0; nt < 2; nt++) {
      #pragma unroll
      for (int kk = 0; kk < 2; kk++) {
        const __bf16* bp = &Kt[nt * 16 + lm][kk * 32 + kh * 8];
        v16bf bf_ = cat16(*(const v8bf*)bp, *(const v8bf*)(bp + 16));
        st[nt] = wmma_bf16(qfrag[kk], bf_, st[nt]);
      }
    }

    // ---- online softmax (per row = r + 8*kh across 16-lane group) ----
    const float scale = 0.125f;  // 1/sqrt(64)
    #pragma unroll
    for (int r = 0; r < 8; r++) {
      const int qrow = qw + r + 8 * kh;
      const int c0 = kv0 + lm;
      const int c1 = c0 + 16;
      float s0 = st[0][r] * scale;
      float s1 = st[1][r] * scale;
      if (c0 > qrow) s0 = -10000.0f;   // matches reference causal mask value
      if (c1 > qrow) s1 = -10000.0f;
      float mx = fmaxf(s0, s1);
      #pragma unroll
      for (int msk = 8; msk >= 1; msk >>= 1) mx = fmaxf(mx, __shfl_xor(mx, msk, 32));
      const float mnew = fmaxf(m_row[r], mx);
      const float alpha = __expf(m_row[r] - mnew);
      const float p0 = __expf(s0 - mnew);
      const float p1 = __expf(s1 - mnew);
      float ps = p0 + p1;
      #pragma unroll
      for (int msk = 8; msk >= 1; msk >>= 1) ps += __shfl_xor(ps, msk, 32);
      l_row[r] = l_row[r] * alpha + ps;
      m_row[r] = mnew;
      #pragma unroll
      for (int dt = 0; dt < 4; dt++) o[dt][r] *= alpha;
      Pb[wave][r + 8 * kh][lm]      = (__bf16)p0;
      Pb[wave][r + 8 * kh][16 + lm] = (__bf16)p1;
    }

    // ---- O += P @ V : A = P (16x32), B = V (32x16 per d-tile) ----
    {
      const __bf16* pp = &Pb[wave][lm][kh * 8];
      v16bf pf = cat16(*(const v8bf*)pp, *(const v8bf*)(pp + 16));
      #pragma unroll
      for (int dt = 0; dt < 4; dt++) {
        const __bf16* vb = &Vt[dt * 16 + lm][kh * 8];
        v16bf bf_ = cat16(*(const v8bf*)vb, *(const v8bf*)(vb + 16));
        o[dt] = wmma_bf16(pf, bf_, o[dt]);
      }
    }
    __syncthreads();
  }

  // ---- epilogue: O /= l, store bf16 to [b,q, h*64+d] ----
  #pragma unroll
  for (int r = 0; r < 8; r++) {
    const float inv = 1.0f / l_row[r];
    const int qrow = qw + r + 8 * kh;
    const size_t base = (size_t)(bb * SS + qrow) * EE + hh * DDIM;
    #pragma unroll
    for (int dt = 0; dt < 4; dt++)
      attn_out[base + dt * 16 + lm] = (__bf16)(o[dt][r] * inv);
  }
}

// ---------------------------------------------------------------------------
// Host orchestration
// ---------------------------------------------------------------------------
extern "C" void kernel_launch(void* const* d_in, const int* in_sizes, int n_in,
                              void* d_out, int out_size, void* d_ws, size_t ws_size,
                              hipStream_t stream) {
  const float* x       = (const float*)d_in[0];
  const float* ln1_g   = (const float*)d_in[1];
  const float* ln1_b   = (const float*)d_in[2];
  const float* w_attn  = (const float*)d_in[3];
  const float* b_attn  = (const float*)d_in[4];
  const float* w_aproj = (const float*)d_in[5];
  const float* b_aproj = (const float*)d_in[6];
  const float* ln2_g   = (const float*)d_in[7];
  const float* ln2_b   = (const float*)d_in[8];
  const float* w_fc    = (const float*)d_in[9];
  const float* b_fc    = (const float*)d_in[10];
  const float* w_mproj = (const float*)d_in[11];
  const float* b_mproj = (const float*)d_in[12];
  float* out = (float*)d_out;

  char* ws = (char*)d_ws;
  auto align256 = [](size_t v) { return (v + 255) & ~(size_t)255; };
  size_t off = 0;
  auto alloc = [&](size_t bytes) { char* p = ws + off; off += align256(bytes); return p; };

  __bf16* wb_attn  = (__bf16*)alloc((size_t)EE * 3 * EE * 2);
  __bf16* wb_aproj = (__bf16*)alloc((size_t)EE * EE * 2);
  __bf16* wb_fc    = (__bf16*)alloc((size_t)EE * 4 * EE * 2);
  __bf16* wb_mproj = (__bf16*)alloc((size_t)4 * EE * EE * 2);
  __bf16* h_ln     = (__bf16*)alloc((size_t)MROWS * EE * 2);
  __bf16* qkv      = (__bf16*)alloc((size_t)MROWS * 3 * EE * 2);
  __bf16* attn_o   = (__bf16*)alloc((size_t)MROWS * EE * 2);
  float*  x1       = (float*) alloc((size_t)MROWS * EE * 4);
  __bf16* h2       = (__bf16*)alloc((size_t)MROWS * EE * 2);
  __bf16* mfc      = (__bf16*)alloc((size_t)MROWS * 4 * EE * 2);
  (void)ws_size; (void)n_in; (void)in_sizes; (void)out_size;

  // 1) weights fp32 -> bf16
  auto cvt = [&](const float* src, __bf16* dst, size_t n) {
    int n4 = (int)(n / 4);
    cvt_f32_bf16_kernel<<<(n4 + 255) / 256, 256, 0, stream>>>(src, dst, n4);
  };
  cvt(w_attn,  wb_attn,  (size_t)EE * 3 * EE);
  cvt(w_aproj, wb_aproj, (size_t)EE * EE);
  cvt(w_fc,    wb_fc,    (size_t)EE * 4 * EE);
  cvt(w_mproj, wb_mproj, (size_t)4 * EE * EE);

  // 2) LN1
  ln_kernel<<<MROWS, 256, 0, stream>>>(x, ln1_g, ln1_b, h_ln);

  // 3) qkv = h_ln @ w_attn + b_attn  -> bf16 [4096, 3072]
  gemm_kernel<0, false, true><<<dim3(3 * EE / 128, MROWS / 128), 256, 0, stream>>>(
      h_ln, wb_attn, b_attn, nullptr, nullptr, qkv, MROWS, 3 * EE, EE);

  // 4) attention
  attn_kernel<<<dim3(SS / 128, BB * HH), 256, 0, stream>>>(qkv, attn_o);

  // 5) x1 = x + attn_o @ w_aproj + b_aproj  -> fp32
  gemm_kernel<0, true, false><<<dim3(EE / 128, MROWS / 128), 256, 0, stream>>>(
      attn_o, wb_aproj, b_aproj, x, x1, nullptr, MROWS, EE, EE);

  // 6) LN2
  ln_kernel<<<MROWS, 256, 0, stream>>>(x1, ln2_g, ln2_b, h2);

  // 7) mfc = gelu(h2 @ w_fc + b_fc) -> bf16 [4096, 4096]
  gemm_kernel<1, false, true><<<dim3(4 * EE / 128, MROWS / 128), 256, 0, stream>>>(
      h2, wb_fc, b_fc, nullptr, nullptr, mfc, MROWS, 4 * EE, EE);

  // 8) out = x1 + mfc @ w_mproj + b_mproj -> fp32
  gemm_kernel<0, true, false><<<dim3(EE / 128, MROWS / 128), 256, 0, stream>>>(
      mfc, wb_mproj, b_mproj, x1, out, nullptr, MROWS, EE, 4 * EE);
}
